// LocallyGroupedSelfAttention_2740189134935
// MI455X (gfx1250) — matrix operand order repaired and checked by reading, not compiled, not run on hardware
//
#include <hip/hip_runtime.h>
#include <hip/hip_bf16.h>

// ---------------------------------------------------------------------------
// LocallyGroupedSelfAttention for MI455X (gfx1250, wave32, WMMA + TDM)
//
// B=512, N=200, C=512, NH=8, HD=64, WS=5  -> M = B*N = 102400 tokens
//   1) cvt_bf16: x, qkv_w, proj_w  fp32 -> bf16 (workspace)
//   2) qkv_attn: block = 80 tokens x 1 head.
//        qkv tile [80x192] = x[80x512] @ qkv_w cols + bias  (WMMA bf16)
//        B tiles staged by Tensor Data Mover (tensor_load_to_lds, 2 LDS bufs)
//        A tiles register-staged (global latency hidden behind WMMA)
//        then 16 windows x (5x5 softmax, P @ V) in LDS, write o bf16
//   3) proj: out = attn_o @ proj_w + proj_b, same TDM/WMMA pipeline
// ---------------------------------------------------------------------------

typedef __bf16        v16bf __attribute__((ext_vector_type(16)));
typedef float         v8f   __attribute__((ext_vector_type(8)));
typedef unsigned int  u32x8 __attribute__((ext_vector_type(8)));
typedef unsigned int  v4u   __attribute__((ext_vector_type(4)));
typedef int           v8i   __attribute__((ext_vector_type(8)));
typedef int           v4i   __attribute__((ext_vector_type(4)));

#define C_    512
#define C3_   1536
#define NH_   8
#define HD_   64
#define WS_   5
#define MTOT  102400
#define SCALE 0.125f

__device__ __forceinline__ unsigned short f2bf(float f) {
    unsigned int u = __float_as_uint(f);
    return (unsigned short)((u + 0x7fffu + ((u >> 16) & 1u)) >> 16);   // RNE
}
__device__ __forceinline__ float bf2f(unsigned short b) {
    return __uint_as_float(((unsigned int)b) << 16);
}
__device__ __forceinline__ unsigned int lds_addr_of(const void* p) {
    // LDS generic address: low 32 bits are the LDS byte offset (ISA 10.2)
    return (unsigned int)(size_t)p;
}

// --- Tensor Data Mover: 2D bf16 tile (tile_w x tile_h elems) -> LDS --------
// D# packing per CDNA5 ISA 8.3/8.4: group0 = {count, lds_addr, global_addr,
// type=2}; group1 = {data_size=2B, tensor dims, tile dims, dim0 stride}.
// 6-arg builtin form (clang-23 / therock-10.0): (v4u, v8i, v4i, v4i, v8i, cpol)
__device__ __forceinline__ void tdm_load_2d_bf16(
    const void* gptr, unsigned int lds_byte_addr,
    unsigned int tensor_w, unsigned int tensor_h,
    unsigned int tile_w, unsigned int tile_h, unsigned int row_stride_elems)
{
    unsigned long long ga = (unsigned long long)(size_t)gptr;
    v4u g0;
    g0.x = 1u;                                                  // count=1 (user D#)
    g0.y = lds_byte_addr;                                       // lds_addr
    g0.z = (unsigned int)ga;                                    // global_addr lo
    g0.w = (unsigned int)((ga >> 32) & 0x01FFFFFFu) | (2u << 30);  // ga hi | type=2
    v8i g1;
    g1[0] = (int)(1u << 16);                                    // data_size=1 -> 2B
    g1[1] = (int)((tensor_w & 0xFFFFu) << 16);                  // tensor_dim0[15:0]
    g1[2] = (int)((tensor_w >> 16) | ((tensor_h & 0xFFFFu) << 16));
    g1[3] = (int)((tensor_h >> 16) | (tile_w << 16));           // tile_dim0
    g1[4] = (int)tile_h;                                        // tile_dim1 (dim2=0)
    g1[5] = (int)row_stride_elems;                              // dim0 stride lo32
    g1[6] = 0;
    g1[7] = 0;
    v4i z4 = {0, 0, 0, 0};                                      // groups 2/3 unused (2D)
    v8i z8 = {0, 0, 0, 0, 0, 0, 0, 0};
    __builtin_amdgcn_tensor_load_to_lds(g0, g1, z4, z4, z8, 0);
}

// ------------------------------ 1) convert ---------------------------------
__global__ __launch_bounds__(256) void cvt_bf16(const float* __restrict__ in,
                                                unsigned int* __restrict__ out,
                                                int n2) {
    int stride = gridDim.x * blockDim.x;
    for (int i = blockIdx.x * blockDim.x + threadIdx.x; i < n2; i += stride) {
        float2 f = ((const float2*)in)[i];
        out[i] = (unsigned int)f2bf(f.x) | ((unsigned int)f2bf(f.y) << 16);
    }
}

// --------------------------- fragment helpers ------------------------------
// A (16-bit 16x32 MxK): lane holds M=lane&15; VGPR v K-pair
//   k0 = (v>=4)*16 + (lane>=16)*8 + (v&3)*2  (K-contiguous in LDS -> b32 reads)
__device__ __forceinline__ v16bf load_a_frag(const unsigned short* row, int lane_hi) {
    u32x8 au;
#pragma unroll
    for (int v = 0; v < 8; ++v) {
        int k0 = ((v & 4) << 2) + lane_hi + ((v & 3) << 1);
        au[v] = *(const unsigned int*)(row + k0);
    }
    return __builtin_bit_cast(v16bf, au);
}
// B (16-bit 32x16 KxN) from row-major [K][ld] tile (TDM layout):
//   lane holds N=lane&15; VGPR v K-pair k0 = (v>=4)*16 + (lane>=16)*8 + (v&3)*2
__device__ __forceinline__ v16bf load_b_frag_rm(const unsigned short* base,
                                                int n, int lane_hi16, int ld) {
    u32x8 bu;
#pragma unroll
    for (int v = 0; v < 8; ++v) {
        int k0 = ((v & 4) << 2) + (lane_hi16 >> 1) + ((v & 3) << 1);
        bu[v] = (unsigned int)base[k0 * ld + n] |
                ((unsigned int)base[(k0 + 1) * ld + n] << 16);
    }
    return __builtin_bit_cast(v16bf, bu);
}

// ----------------------- 2) fused QKV GEMM + attention ---------------------
#define TG_M  80
#define TG_K  32
#define QKVN  192

__global__ __launch_bounds__(256) void qkv_attn(
    const unsigned short* __restrict__ xb,   // [MTOT][512]  bf16
    const unsigned short* __restrict__ wq,   // [512][1536]  bf16
    const float*          __restrict__ qb,   // [1536]
    unsigned short*       __restrict__ ob)   // [MTOT][512]  bf16
{
    __shared__ unsigned short As[TG_M][TG_K];          // 5.0 KB
    __shared__ unsigned short Bs[2][3][TG_K][64];      // 24 KB, TDM dest (row-major)
    __shared__ unsigned short S[TG_M][QKVN + 4];       // 30.6 KB q|k|v bf16
    __shared__ float          SC[16][28];              // 1.8 KB

    const int tg   = blockIdx.x;
    const int h    = blockIdx.y;
    const int tid  = threadIdx.x;
    const int wid  = tid >> 5;
    const int lane = tid & 31;
    const int row0 = tg * TG_M;

    const int cg  = wid & 3;       // col tiles {3cg..3cg+2} of 12
    const int wr  = wid >> 2;      // wr=0 -> rt{0,1,2}, wr=1 -> rt{3,4}
    const int nrt = wr ? 2 : 3;

    v8f acc[3][3];
#pragma unroll
    for (int a = 0; a < 3; ++a)
#pragma unroll
        for (int b = 0; b < 3; ++b) acc[a][b] = (v8f)0.0f;

    const int a_hi = (lane >> 4) << 3;
    const int b_hi = (lane >> 4) << 4;
    const int Mrow = lane & 15;

    // ---- prologue: stage chunk 0 (TDM for B, registers->LDS for A) ----
    if (tid == 0) {
#pragma unroll
        for (int g = 0; g < 3; ++g)
            tdm_load_2d_bf16(wq + (size_t)g * C_ + (h << 6),
                             lds_addr_of(&Bs[0][g][0][0]),
                             C3_, C_, 64, TG_K, C3_);
    }
    {
        unsigned int areg[5];
#pragma unroll
        for (int t = 0; t < 5; ++t) {
            int i = tid + t * 256, m = i >> 4, k2 = i & 15;
            areg[t] = *(const unsigned int*)(xb + (size_t)(row0 + m) * C_ + (k2 << 1));
        }
#pragma unroll
        for (int t = 0; t < 5; ++t) {
            int i = tid + t * 256, m = i >> 4, k2 = i & 15;
            ((unsigned int*)&As[m][0])[k2] = areg[t];
        }
    }
    if (tid == 0) __builtin_amdgcn_s_wait_tensorcnt(0);
    __syncthreads();

    // ---- K loop, software pipelined ----
    for (int kk = 0; kk < C_; kk += TG_K) {
        const int  buf  = (kk >> 5) & 1;
        const bool next = (kk + TG_K) < C_;

        unsigned int areg[5];
        if (next) {
            if (tid == 0) {    // TDM ignores EXEC; one wave issues 3 tiles
#pragma unroll
                for (int g = 0; g < 3; ++g)
                    tdm_load_2d_bf16(wq + (size_t)(kk + TG_K) * C3_ + g * C_ + (h << 6),
                                     lds_addr_of(&Bs[buf ^ 1][g][0][0]),
                                     C3_, C_, 64, TG_K, C3_);
            }
#pragma unroll
            for (int t = 0; t < 5; ++t) {
                int i = tid + t * 256, m = i >> 4, k2 = i & 15;
                areg[t] = *(const unsigned int*)(xb + (size_t)(row0 + m) * C_ +
                                                 (kk + TG_K) + (k2 << 1));
            }
        }

        // compute on current buffers while next-chunk loads are in flight
        v16bf bfrag[3];
#pragma unroll
        for (int j = 0; j < 3; ++j) {
            int ct = cg * 3 + j;
            bfrag[j] = load_b_frag_rm(&Bs[buf][ct >> 2][0][(ct & 3) << 4],
                                      Mrow, b_hi, 64);
        }
#pragma unroll
        for (int r = 0; r < 3; ++r) {
            if (r < nrt) {
                int rt = wr * 3 + r;
                v16bf afrag = load_a_frag(&As[rt * 16 + Mrow][0], a_hi);
#pragma unroll
                for (int j = 0; j < 3; ++j)
                    acc[r][j] = __builtin_amdgcn_wmma_f32_16x16x32_bf16(
                        false, afrag, false, bfrag[j], (short)0, acc[r][j],
                        false, false);
            }
        }
        __syncthreads();               // all waves done reading As / Bs[buf]

        if (next) {
#pragma unroll
            for (int t = 0; t < 5; ++t) {
                int i = tid + t * 256, m = i >> 4, k2 = i & 15;
                ((unsigned int*)&As[m][0])[k2] = areg[t];
            }
            if (tid == 0) __builtin_amdgcn_s_wait_tensorcnt(0);
        }
        __syncthreads();
    }

    // ---- epilogue: + bias, q|k|v -> LDS bf16 ----
#pragma unroll
    for (int r = 0; r < 3; ++r) {
        if (r < nrt) {
            int rt = wr * 3 + r;
#pragma unroll
            for (int j = 0; j < 3; ++j) {
                int col  = (cg * 3 + j) * 16 + Mrow;
                float bv = qb[((col >> 6) << 9) + (h << 6) + (col & 63)];
                int rb   = rt * 16 + a_hi;
#pragma unroll
                for (int i = 0; i < 8; ++i)
                    S[rb + i][col] = f2bf(acc[r][j][i] + bv);
            }
        }
    }
    __syncthreads();

    // ---- attention: 16 windows x 16 threads ----
    const int wg = tid >> 4, wt = tid & 15;
    const int r0 = wg * WS_;
    for (int p = wt; p < WS_ * WS_; p += 16) {
        int i = p / WS_, j = p % WS_;
        float s = 0.f;
#pragma unroll
        for (int d = 0; d < HD_; ++d)
            s += bf2f(S[r0 + i][d]) * bf2f(S[r0 + j][HD_ + d]);
        SC[wg][p] = s * SCALE;
    }
    __syncthreads();

    const int d0 = wt << 2;
#pragma unroll
    for (int i = 0; i < WS_; ++i) {
        float sc[WS_], mx = -3.4e38f;
#pragma unroll
        for (int j = 0; j < WS_; ++j) { sc[j] = SC[wg][i * WS_ + j]; mx = fmaxf(mx, sc[j]); }
        float e[WS_], sum = 0.f;
#pragma unroll
        for (int j = 0; j < WS_; ++j) { e[j] = __expf(sc[j] - mx); sum += e[j]; }
        float inv = 1.f / sum;
        float o0 = 0.f, o1 = 0.f, o2 = 0.f, o3 = 0.f;
#pragma unroll
        for (int j = 0; j < WS_; ++j) {
            float pw = e[j] * inv;
            o0 += pw * bf2f(S[r0 + j][128 + d0 + 0]);
            o1 += pw * bf2f(S[r0 + j][128 + d0 + 1]);
            o2 += pw * bf2f(S[r0 + j][128 + d0 + 2]);
            o3 += pw * bf2f(S[r0 + j][128 + d0 + 3]);
        }
        unsigned long long pack =
            (unsigned long long)f2bf(o0)         | ((unsigned long long)f2bf(o1) << 16) |
            ((unsigned long long)f2bf(o2) << 32) | ((unsigned long long)f2bf(o3) << 48);
        *(unsigned long long*)(ob + (size_t)(row0 + r0 + i) * C_ + (h << 6) + d0) = pack;
    }
}

// ------------------------------ 3) projection ------------------------------
#define PM 128

__global__ __launch_bounds__(256) void proj(
    const unsigned short* __restrict__ ab,   // [MTOT][512] bf16
    const unsigned short* __restrict__ pw,   // [512][512]  bf16
    const float*          __restrict__ pb,   // [512]
    float*                __restrict__ out)  // [MTOT][512] fp32
{
    __shared__ unsigned short As[PM][TG_K];        // 8 KB
    __shared__ unsigned short Bs[2][TG_K][64];     // 8 KB, TDM dest

    const int tid  = threadIdx.x;
    const int wid  = tid >> 5;
    const int lane = tid & 31;
    const int row0 = blockIdx.x * PM;
    const int col0 = blockIdx.y * 64;
    const int a_hi = (lane >> 4) << 3;
    const int b_hi = (lane >> 4) << 4;
    const int Mrow = lane & 15;

    v8f acc[4];
#pragma unroll
    for (int j = 0; j < 4; ++j) acc[j] = (v8f)0.0f;

    // prologue
    if (tid == 0)
        tdm_load_2d_bf16(pw + col0, lds_addr_of(&Bs[0][0][0]),
                         C_, C_, 64, TG_K, C_);
    {
        unsigned int areg[8];
#pragma unroll
        for (int t = 0; t < 8; ++t) {
            int i = tid + t * 256, m = i >> 4, k2 = i & 15;
            areg[t] = *(const unsigned int*)(ab + (size_t)(row0 + m) * C_ + (k2 << 1));
        }
#pragma unroll
        for (int t = 0; t < 8; ++t) {
            int i = tid + t * 256, m = i >> 4, k2 = i & 15;
            ((unsigned int*)&As[m][0])[k2] = areg[t];
        }
    }
    if (tid == 0) __builtin_amdgcn_s_wait_tensorcnt(0);
    __syncthreads();

    for (int kk = 0; kk < C_; kk += TG_K) {
        const int  buf  = (kk >> 5) & 1;
        const bool next = (kk + TG_K) < C_;

        unsigned int areg[8];
        if (next) {
            if (tid == 0)
                tdm_load_2d_bf16(pw + (size_t)(kk + TG_K) * C_ + col0,
                                 lds_addr_of(&Bs[buf ^ 1][0][0]),
                                 C_, C_, 64, TG_K, C_);
#pragma unroll
            for (int t = 0; t < 8; ++t) {
                int i = tid + t * 256, m = i >> 4, k2 = i & 15;
                areg[t] = *(const unsigned int*)(ab + (size_t)(row0 + m) * C_ +
                                                 (kk + TG_K) + (k2 << 1));
            }
        }

        v16bf afrag = load_a_frag(&As[wid * 16 + Mrow][0], a_hi);
#pragma unroll
        for (int j = 0; j < 4; ++j) {
            v16bf bfrag = load_b_frag_rm(&Bs[buf][0][j << 4], Mrow, b_hi, 64);
            acc[j] = __builtin_amdgcn_wmma_f32_16x16x32_bf16(
                false, afrag, false, bfrag, (short)0, acc[j], false, false);
        }
        __syncthreads();

        if (next) {
#pragma unroll
            for (int t = 0; t < 8; ++t) {
                int i = tid + t * 256, m = i >> 4, k2 = i & 15;
                ((unsigned int*)&As[m][0])[k2] = areg[t];
            }
            if (tid == 0) __builtin_amdgcn_s_wait_tensorcnt(0);
        }
        __syncthreads();
    }

#pragma unroll
    for (int j = 0; j < 4; ++j) {
        int col  = col0 + j * 16 + Mrow;
        float bv = pb[col];
        int rb   = row0 + wid * 16 + a_hi;
#pragma unroll
        for (int i = 0; i < 8; ++i)
            out[(size_t)(rb + i) * C_ + col] = acc[j][i] + bv;
    }
}

// ------------------------------- launcher ----------------------------------
extern "C" void kernel_launch(void* const* d_in, const int* in_sizes, int n_in,
                              void* d_out, int out_size, void* d_ws, size_t ws_size,
                              hipStream_t stream) {
    (void)in_sizes; (void)n_in; (void)out_size; (void)ws_size;

    const float* x      = (const float*)d_in[0];
    const float* qkv_w  = (const float*)d_in[1];
    const float* qkv_b  = (const float*)d_in[2];
    const float* proj_w = (const float*)d_in[3];
    const float* proj_b = (const float*)d_in[4];
    float* out          = (float*)d_out;

    char* w = (char*)d_ws;                                        // ~211 MB
    unsigned short* xb  = (unsigned short*)w;
    unsigned short* wqb = (unsigned short*)(w + 104857600);
    unsigned short* wpb = (unsigned short*)(w + 104857600 + 1572864);
    unsigned short* ob  = (unsigned short*)(w + 104857600 + 1572864 + 524288);

    cvt_bf16<<<4096, 256, 0, stream>>>(x,      (unsigned int*)xb,  (MTOT * C_) / 2);
    cvt_bf16<<<512,  256, 0, stream>>>(qkv_w,  (unsigned int*)wqb, (C_ * C3_) / 2);
    cvt_bf16<<<256,  256, 0, stream>>>(proj_w, (unsigned int*)wpb, (C_ * C_) / 2);

    qkv_attn<<<dim3(MTOT / TG_M, NH_), 256, 0, stream>>>(xb, wqb, qkv_b, ob);
    proj    <<<dim3(MTOT / PM,   C_ / 64), 256, 0, stream>>>(ob, wpb, proj_b, out);
}
